// RG_SA_2680059592816
// MI455X (gfx1250) — compile-verified, matrix operand order checked by reading, not run on hardware
//
#include <hip/hip_runtime.h>
#include <hip/hip_bf16.h>

// ---------------------------------------------------------------------------
// RG-SA block for MI455X (gfx1250, wave32, WMMA).
// All large GEMMs (fuse-conv-as-GEMM, q, k, v, attention scores, attention*V,
// projection) run on v_wmma_f32_16x16x32_f16.  Attention is flash-style
// streamed.  WMMA fragments are loaded as 2x128-bit vector loads per operand
// (ISA 7.12.2 per-lane layout is two contiguous 16B chunks), with weights and
// V pre-transposed so every operand is per-lane row-major.
// ---------------------------------------------------------------------------

typedef __attribute__((ext_vector_type(16))) _Float16 v16h;
typedef __attribute__((ext_vector_type(8)))  _Float16 v8h;
typedef __attribute__((ext_vector_type(8)))  float    v8f;

#define B_   2
#define H_   128
#define W_   128
#define C_   384
#define N_   (H_*W_)        // 16384
#define CR_  192
#define CG_  128
#define L_   1024           // (H/4)*(W/4)
#define NH_  8
#define DK_  24             // cr/heads
#define DKP_ 32             // padded K for WMMA
#define HD_  48             // C/heads
#define KF_  6144           // 384*4*4 im2col K
#define SCALE_ 0.20412414523193154f   // 24^-0.5

// 16-bit A/B fragment (16x32): lane holds K[8h..8h+7] and K[16+8h..16+8h+7].
__device__ __forceinline__ v16h frag_from_f16(const _Float16* row, int half) {
  const v8h* p = (const v8h*)row;
  v8h lo = p[half];
  v8h hi = p[2 + half];
  v16h o;
#pragma unroll
  for (int i = 0; i < 8; ++i) { o[i] = lo[i]; o[8 + i] = hi[i]; }
  return o;
}

__device__ __forceinline__ v16h frag_from_f32(const float* row, int half) {
  const float4* p = (const float4*)row;
  float4 a0 = p[2 * half], a1 = p[2 * half + 1];
  float4 b0 = p[4 + 2 * half], b1 = p[5 + 2 * half];
  v16h o;
  o[0]=(_Float16)a0.x; o[1]=(_Float16)a0.y; o[2]=(_Float16)a0.z; o[3]=(_Float16)a0.w;
  o[4]=(_Float16)a1.x; o[5]=(_Float16)a1.y; o[6]=(_Float16)a1.z; o[7]=(_Float16)a1.w;
  o[8]=(_Float16)b0.x; o[9]=(_Float16)b0.y; o[10]=(_Float16)b0.z; o[11]=(_Float16)b0.w;
  o[12]=(_Float16)b1.x; o[13]=(_Float16)b1.y; o[14]=(_Float16)b1.z; o[15]=(_Float16)b1.w;
  return o;
}

// ---------------------------------------------------------------------------
// Generic depthwise conv over [B, CG, H, W].
// ---------------------------------------------------------------------------
__global__ void k_dwconv(const float* __restrict__ src, float* __restrict__ dst,
                         const float* __restrict__ w, const float* __restrict__ bias,
                         int ks, int pad, int src_chlast, int src_cbase, int src_ctot,
                         int dst_cbase, int dst_ctot) {
  int t = blockIdx.x * blockDim.x + threadIdx.x;
  if (t >= B_ * CG_ * H_ * W_) return;
  int xw = t & (W_ - 1);
  int y  = (t >> 7) & (H_ - 1);
  int c  = (t >> 14) & (CG_ - 1);
  int b  = t >> 21;
  float acc = bias[c];
  for (int i = 0; i < ks; ++i) {
    int yy = y + i - pad;
    if (yy < 0 || yy >= H_) continue;
    for (int j = 0; j < ks; ++j) {
      int xx = xw + j - pad;
      if (xx < 0 || xx >= W_) continue;
      float s;
      if (src_chlast)
        s = src[((size_t)b * N_ + yy * W_ + xx) * C_ + src_cbase + c];
      else
        s = src[(((size_t)b * src_ctot + c) * H_ + yy) * W_ + xx];
      acc += s * w[(c * ks + i) * ks + j];
    }
  }
  dst[(((size_t)b * dst_ctot + dst_cbase + c) * H_ + y) * W_ + xw] = acc;
}

// im2col for the 4x4/stride-4 conv: PF[row=b*L+l][idx=cin*16+i*4+j] (f16)
__global__ void k_im2col(const float* __restrict__ fused, _Float16* __restrict__ pf) {
  int t = blockIdx.x * blockDim.x + threadIdx.x;
  if (t >= B_ * L_ * KF_) return;
  int idx = t % KF_, row = t / KF_;
  int b = row >> 10, l = row & (L_ - 1);
  int y = l >> 5, xw = l & 31;
  int cin = idx >> 4, rem = idx & 15, i = rem >> 2, j = rem & 3;
  pf[(size_t)row * KF_ + idx] =
      (_Float16)fused[(((size_t)b * C_ + cin) * H_ + 4 * y + i) * W_ + 4 * xw + j];
}

__global__ void k_cvt_f16(const float* __restrict__ s, _Float16* __restrict__ d, int n) {
  int t = blockIdx.x * blockDim.x + threadIdx.x;
  if (t < n) d[t] = (_Float16)s[t];
}

// transpose-convert: src [K,Nn] f32 -> dst [Nn,K] f16
__global__ void k_cvt_t(const float* __restrict__ src, _Float16* __restrict__ dst,
                        int K, int Nn) {
  int t = blockIdx.x * blockDim.x + threadIdx.x;
  if (t >= K * Nn) return;
  int k = t / Nn, n = t % Nn;
  dst[(size_t)n * K + k] = (_Float16)src[t];
}

// ---------------------------------------------------------------------------
// fuse conv as GEMM: PF [2048,6144] x fus_w_f16(N-major) [192,6144] + GELU
// ---------------------------------------------------------------------------
__global__ void k_gemm_fus(const _Float16* __restrict__ pf, const _Float16* __restrict__ fw,
                           const float* __restrict__ fb, _Float16* __restrict__ xr) {
  int lane = threadIdx.x, mn = lane & 15, half = lane >> 4;
  int mtile = blockIdx.x, ntile = blockIdx.y;
  int arow = mtile * 16 + mn, ncol = ntile * 16 + mn;
  v8f acc = {0.f,0.f,0.f,0.f,0.f,0.f,0.f,0.f};
  for (int kb = 0; kb < KF_; kb += 32) {
    v16h a  = frag_from_f16(pf + (size_t)arow * KF_ + kb, half);
    v16h bm = frag_from_f16(fw + (size_t)ncol * KF_ + kb, half);
    acc = __builtin_amdgcn_wmma_f32_16x16x32_f16(false, a, false, bm, (short)0, acc, false, false);
  }
  float bias = fb[ncol];
#pragma unroll
  for (int r = 0; r < 8; ++r) {
    int grow = mtile * 16 + r + 8 * half;
    float v = acc[r] + bias;
    float g = 0.5f * v * (1.0f + erff(v * 0.70710678118654752f));
    xr[(size_t)grow * CR_ + ncol] = (_Float16)g;
  }
}

// q = x @ q_w (A f32, B = qwT [192,384] f16) -> q_pad f16 [B,8,N,32]
__global__ void k_gemm_q(const float* __restrict__ x, const _Float16* __restrict__ qwt,
                         _Float16* __restrict__ qp) {
  int lane = threadIdx.x, mn = lane & 15, half = lane >> 4;
  int mtile = blockIdx.x, ntile = blockIdx.y;
  int arow = mtile * 16 + mn, ncol = ntile * 16 + mn;
  v8f acc = {0.f,0.f,0.f,0.f,0.f,0.f,0.f,0.f};
  for (int kb = 0; kb < C_; kb += 32) {
    v16h a  = frag_from_f32(x + (size_t)arow * C_ + kb, half);
    v16h bm = frag_from_f16(qwt + (size_t)ncol * C_ + kb, half);
    acc = __builtin_amdgcn_wmma_f32_16x16x32_f16(false, a, false, bm, (short)0, acc, false, false);
  }
  int head = ncol / DK_, kk = ncol % DK_;
#pragma unroll
  for (int r = 0; r < 8; ++r) {
    int grow = mtile * 16 + r + 8 * half;
    int b = grow >> 14, nrow = grow & (N_ - 1);
    qp[(((size_t)(b * NH_ + head) * N_ + nrow) * DKP_) + kk] = (_Float16)acc[r];
  }
}

// k = _xr @ k_w (B = kwT [192,192] f16) -> k_pad f16 [B,8,L,32]
__global__ void k_gemm_k(const _Float16* __restrict__ xr, const _Float16* __restrict__ kwt,
                         _Float16* __restrict__ kp) {
  int lane = threadIdx.x, mn = lane & 15, half = lane >> 4;
  int mtile = blockIdx.x, ntile = blockIdx.y;
  int arow = mtile * 16 + mn, ncol = ntile * 16 + mn;
  v8f acc = {0.f,0.f,0.f,0.f,0.f,0.f,0.f,0.f};
  for (int kb = 0; kb < CR_; kb += 32) {
    v16h a  = frag_from_f16(xr + (size_t)arow * CR_ + kb, half);
    v16h bm = frag_from_f16(kwt + (size_t)ncol * CR_ + kb, half);
    acc = __builtin_amdgcn_wmma_f32_16x16x32_f16(false, a, false, bm, (short)0, acc, false, false);
  }
  int head = ncol / DK_, kk = ncol % DK_;
#pragma unroll
  for (int r = 0; r < 8; ++r) {
    int grow = mtile * 16 + r + 8 * half;
    int b = grow >> 10, lrow = grow & (L_ - 1);
    kp[(((size_t)(b * NH_ + head) * L_ + lrow) * DKP_) + kk] = (_Float16)acc[r];
  }
}

// v = _xr @ v_w (B = vwT [384,192] f16) -> vbuf f32 [B,L,384]
__global__ void k_gemm_v(const _Float16* __restrict__ xr, const _Float16* __restrict__ vwt,
                         float* __restrict__ vbuf) {
  int lane = threadIdx.x, mn = lane & 15, half = lane >> 4;
  int mtile = blockIdx.x, ntile = blockIdx.y;
  int arow = mtile * 16 + mn, ncol = ntile * 16 + mn;
  v8f acc = {0.f,0.f,0.f,0.f,0.f,0.f,0.f,0.f};
  for (int kb = 0; kb < CR_; kb += 32) {
    v16h a  = frag_from_f16(xr + (size_t)arow * CR_ + kb, half);
    v16h bm = frag_from_f16(vwt + (size_t)ncol * CR_ + kb, half);
    acc = __builtin_amdgcn_wmma_f32_16x16x32_f16(false, a, false, bm, (short)0, acc, false, false);
  }
#pragma unroll
  for (int r = 0; r < 8; ++r) {
    int grow = mtile * 16 + r + 8 * half;
    vbuf[(size_t)grow * C_ + ncol] = acc[r];
  }
}

// v += cpe(dwconv3x3 over [B,384,32,32]); emit V TRANSPOSED f16 [B,8,48,L]
__global__ void k_cpe_add(const float* __restrict__ vbuf, const float* __restrict__ cw,
                          const float* __restrict__ cb, _Float16* __restrict__ vt) {
  int t = blockIdx.x * blockDim.x + threadIdx.x;
  if (t >= B_ * L_ * C_) return;
  int c = t % C_;
  int l = (t / C_) % L_;
  int b = t / (C_ * L_);
  int y = l >> 5, xw = l & 31;
  float acc = cb[c];
  for (int i = 0; i < 3; ++i) {
    int yy = y + i - 1;
    if (yy < 0 || yy >= 32) continue;
    for (int j = 0; j < 3; ++j) {
      int xx = xw + j - 1;
      if (xx < 0 || xx >= 32) continue;
      acc += vbuf[((size_t)b * L_ + yy * 32 + xx) * C_ + c] * cw[(c * 3 + i) * 3 + j];
    }
  }
  float vnew = vbuf[((size_t)b * L_ + l) * C_ + c] + acc;
  int h = c / HD_, d = c % HD_;
  vt[(((size_t)(b * NH_ + h) * HD_ + d) * L_) + l] = (_Float16)vnew;
}

// ---------------------------------------------------------------------------
// Flash attention: one wave per (b, head, 16-row query tile); 32 keys/iter.
// All fragments loaded as 2x128b; P relayout via 1KB LDS (ds_load_b128 back).
// ---------------------------------------------------------------------------
__global__ void k_attn(const _Float16* __restrict__ qp, const _Float16* __restrict__ kp,
                       const _Float16* __restrict__ vt, float* __restrict__ out) {
  __shared__ __attribute__((aligned(16))) _Float16 lds_p[16 * 32];
  int lane = threadIdx.x, mn = lane & 15, half = lane >> 4;
  int mtile = blockIdx.x;          // 0..1023
  int h = blockIdx.y;              // 0..7
  int b = blockIdx.z;              // 0..1
  int bh = b * NH_ + h;

  v16h qa = frag_from_f16(qp + ((size_t)bh * N_ + mtile * 16 + mn) * DKP_, half);

  float rm[8], rs[8];
#pragma unroll
  for (int r = 0; r < 8; ++r) { rm[r] = -1e30f; rs[r] = 0.f; }
  v8f o0 = {0.f,0.f,0.f,0.f,0.f,0.f,0.f,0.f};
  v8f o1 = o0, o2 = o0;

  const _Float16* kbase = kp + (size_t)bh * L_ * DKP_;
  const _Float16* vbase0 = vt + ((size_t)bh * HD_ + mn) * L_;
  const _Float16* vbase1 = vt + ((size_t)bh * HD_ + 16 + mn) * L_;
  const _Float16* vbase2 = vt + ((size_t)bh * HD_ + 32 + mn) * L_;

  for (int base = 0; base < L_; base += 32) {
    v16h kb0 = frag_from_f16(kbase + (size_t)(base + mn) * DKP_, half);
    v16h kb1 = frag_from_f16(kbase + (size_t)(base + 16 + mn) * DKP_, half);
    v8f z = {0.f,0.f,0.f,0.f,0.f,0.f,0.f,0.f};
    v8f s0 = __builtin_amdgcn_wmma_f32_16x16x32_f16(false, qa, false, kb0, (short)0, z, false, false);
    v8f s1 = __builtin_amdgcn_wmma_f32_16x16x32_f16(false, qa, false, kb1, (short)0, z, false, false);

#pragma unroll
    for (int r = 0; r < 8; ++r) {
      float a0 = s0[r] * SCALE_, a1 = s1[r] * SCALE_;
      float cm = fmaxf(a0, a1);
      cm = fmaxf(cm, __shfl_xor(cm, 1, 32));
      cm = fmaxf(cm, __shfl_xor(cm, 2, 32));
      cm = fmaxf(cm, __shfl_xor(cm, 4, 32));
      cm = fmaxf(cm, __shfl_xor(cm, 8, 32));
      float mnew = fmaxf(rm[r], cm);
      float alpha = __expf(rm[r] - mnew);
      float p0 = __expf(a0 - mnew);
      float p1 = __expf(a1 - mnew);
      float ps = p0 + p1;
      ps += __shfl_xor(ps, 1, 32);
      ps += __shfl_xor(ps, 2, 32);
      ps += __shfl_xor(ps, 4, 32);
      ps += __shfl_xor(ps, 8, 32);
      rs[r] = rs[r] * alpha + ps;
      rm[r] = mnew;
      o0[r] *= alpha; o1[r] *= alpha; o2[r] *= alpha;
      lds_p[(8 * half + r) * 32 + mn]      = (_Float16)p0;
      lds_p[(8 * half + r) * 32 + 16 + mn] = (_Float16)p1;
    }
    __syncthreads();
    v16h pa = frag_from_f16(lds_p + mn * 32, half);
    __syncthreads();

    v16h vb0 = frag_from_f16(vbase0 + base, half);
    v16h vb1 = frag_from_f16(vbase1 + base, half);
    v16h vb2 = frag_from_f16(vbase2 + base, half);
    o0 = __builtin_amdgcn_wmma_f32_16x16x32_f16(false, pa, false, vb0, (short)0, o0, false, false);
    o1 = __builtin_amdgcn_wmma_f32_16x16x32_f16(false, pa, false, vb1, (short)0, o1, false, false);
    o2 = __builtin_amdgcn_wmma_f32_16x16x32_f16(false, pa, false, vb2, (short)0, o2, false, false);
  }

#pragma unroll
  for (int r = 0; r < 8; ++r) {
    float inv = 1.0f / rs[r];
    int nrow = mtile * 16 + r + 8 * half;
    float* op = out + ((size_t)b * N_ + nrow) * C_ + h * HD_;
    op[mn]      = o0[r] * inv;
    op[16 + mn] = o1[r] * inv;
    op[32 + mn] = o2[r] * inv;
  }
}

// out = out_attn @ proj_w + proj_b (A f32, B = pwT [384,384] f16)
__global__ void k_gemm_proj(const float* __restrict__ oa, const _Float16* __restrict__ pwt,
                            const float* __restrict__ pb, float* __restrict__ out) {
  int lane = threadIdx.x, mn = lane & 15, half = lane >> 4;
  int mtile = blockIdx.x, ntile = blockIdx.y;
  int arow = mtile * 16 + mn, ncol = ntile * 16 + mn;
  v8f acc = {0.f,0.f,0.f,0.f,0.f,0.f,0.f,0.f};
  for (int kb = 0; kb < C_; kb += 32) {
    v16h a  = frag_from_f32(oa + (size_t)arow * C_ + kb, half);
    v16h bm = frag_from_f16(pwt + (size_t)ncol * C_ + kb, half);
    acc = __builtin_amdgcn_wmma_f32_16x16x32_f16(false, a, false, bm, (short)0, acc, false, false);
  }
  float bias = pb[ncol];
#pragma unroll
  for (int r = 0; r < 8; ++r) {
    int grow = mtile * 16 + r + 8 * half;
    out[(size_t)grow * C_ + ncol] = acc[r] + bias;
  }
}

// ---------------------------------------------------------------------------
extern "C" void kernel_launch(void* const* d_in, const int* in_sizes, int n_in,
                              void* d_out, int out_size, void* d_ws, size_t ws_size,
                              hipStream_t stream) {
  const float* x      = (const float*)d_in[0];
  const float* red_w1 = (const float*)d_in[1];
  const float* red_b1 = (const float*)d_in[2];
  const float* red_w3 = (const float*)d_in[3];
  const float* red_b3 = (const float*)d_in[4];
  const float* red_w5 = (const float*)d_in[5];
  const float* red_b5 = (const float*)d_in[6];
  const float* fus_w  = (const float*)d_in[7];
  const float* fus_b  = (const float*)d_in[8];
  const float* q_w    = (const float*)d_in[9];
  const float* k_w    = (const float*)d_in[10];
  const float* v_w    = (const float*)d_in[11];
  const float* cpe_w  = (const float*)d_in[12];
  const float* cpe_b  = (const float*)d_in[13];
  const float* proj_w = (const float*)d_in[14];
  const float* proj_b = (const float*)d_in[15];
  float* out = (float*)d_out;

  char* w = (char*)d_ws;
  // Workspace layout (~88 MB, with reuse):
  float*    FUSED = (float*)(w + 0);                  // 50331648 B; reused as OUT_ATTN
  float*    TMPA  = (float*)(w + 50331648);           // 16777216 B
  char*     TMPBc = w + 67108864;                     // 16777216 B
  _Float16* XRH   = (_Float16*)(w + 83886080);        // 786432 B
  _Float16* WTQ   = (_Float16*)(w + 84672512);        // 147456 B  (qwT [192,384])
  _Float16* WTK   = (_Float16*)(w + 84819968);        // 73728 B   (kwT [192,192])
  _Float16* WTV   = (_Float16*)(w + 84893696);        // 147456 B  (vwT [384,192])
  _Float16* WTP   = (_Float16*)(w + 85041152);        // 294912 B  (pwT [384,384])
  _Float16* FWH   = (_Float16*)(w + 85336064);        // 2359296 B (fus_w f16, N-major)

  float*    TMPB  = (float*)TMPBc;
  _Float16* PF    = (_Float16*)TMPA;                  // im2col, 25165824 B (spans TMPA+TMPB)
  _Float16* QPAD  = (_Float16*)TMPA;                  // later: 16777216 B
  _Float16* KPAD  = (_Float16*)TMPBc;                 // later: 1048576 B
  float*    VBUF  = (float*)(TMPBc + 1048576);        // later: 3145728 B
  _Float16* VT    = (_Float16*)(TMPBc + 4194304);     // later: 1572864 B (V transposed)
  float*    OATT  = FUSED;

  const int dwN = B_ * CG_ * H_ * W_;
  dim3 blk(256);
  dim3 dwg((dwN + 255) / 256);

  // --- depthwise reduction branches -> FUSED [B,384,128,128] (planar)
  k_dwconv<<<dwg, blk, 0, stream>>>(x, FUSED, red_w1, red_b1, 1, 0, 1, 0,   0,   0,   C_);
  k_dwconv<<<dwg, blk, 0, stream>>>(x, TMPA,  red_w3, red_b3, 3, 1, 1, 128, 0,   0,   CG_);
  k_dwconv<<<dwg, blk, 0, stream>>>(TMPA, FUSED, red_w3, red_b3, 3, 1, 0, 0, CG_, 128, C_);
  k_dwconv<<<dwg, blk, 0, stream>>>(x, TMPB,  red_w5, red_b5, 5, 2, 1, 256, 0,   0,   CG_);
  k_dwconv<<<dwg, blk, 0, stream>>>(TMPB, TMPA, red_w5, red_b5, 5, 2, 0, 0, CG_, 0,   CG_);
  k_dwconv<<<dwg, blk, 0, stream>>>(TMPA, FUSED, red_w5, red_b5, 5, 2, 0, 0, CG_, 256, C_);

  // --- weight conversions (transposed to N-major f16)
  k_cvt_t<<<(C_*CR_ + 255)/256, blk, 0, stream>>>(q_w, WTQ, C_, CR_);
  k_cvt_t<<<(CR_*CR_ + 255)/256, blk, 0, stream>>>(k_w, WTK, CR_, CR_);
  k_cvt_t<<<(CR_*C_ + 255)/256, blk, 0, stream>>>(v_w, WTV, CR_, C_);
  k_cvt_t<<<(C_*C_ + 255)/256, blk, 0, stream>>>(proj_w, WTP, C_, C_);
  k_cvt_f16<<<(CR_*KF_ + 255)/256, blk, 0, stream>>>(fus_w, FWH, CR_*KF_);

  // --- fuse conv: im2col + WMMA GEMM + GELU -> _xr f16 [2048,192]
  k_im2col<<<((B_*L_*KF_) + 255)/256, blk, 0, stream>>>(FUSED, PF);
  k_gemm_fus<<<dim3((B_*L_)/16, CR_/16), dim3(32), 0, stream>>>(PF, FWH, fus_b, XRH);

  // --- q GEMM (zero-pad dk 24->32 first; QPAD overlays PF which is now dead)
  hipMemsetAsync(QPAD, 0, (size_t)B_*NH_*N_*DKP_*2, stream);
  k_gemm_q<<<dim3((B_*N_)/16, CR_/16), dim3(32), 0, stream>>>(x, WTQ, QPAD);

  // --- k GEMM (padded), v GEMM, CPE (emits V transposed)
  hipMemsetAsync(KPAD, 0, (size_t)B_*NH_*L_*DKP_*2, stream);
  k_gemm_k<<<dim3((B_*L_)/16, CR_/16), dim3(32), 0, stream>>>(XRH, WTK, KPAD);
  k_gemm_v<<<dim3((B_*L_)/16, C_/16),  dim3(32), 0, stream>>>(XRH, WTV, VBUF);
  k_cpe_add<<<(B_*L_*C_ + 255)/256, blk, 0, stream>>>(VBUF, cpe_w, cpe_b, VT);

  // --- flash attention -> OATT (reuses FUSED)
  k_attn<<<dim3(N_/16, NH_, B_), dim3(32), 0, stream>>>(QPAD, KPAD, VT, OATT);

  // --- projection
  k_gemm_proj<<<dim3((B_*N_)/16, C_/16), dim3(32), 0, stream>>>(OATT, WTP, proj_b, out);
}
